// NLFModel_82291573391854
// MI455X (gfx1250) — matrix operand order repaired and checked by reading, not compiled, not run on hardware
//
#include <hip/hip_runtime.h>
#include <hip/hip_bf16.h>

typedef __attribute__((ext_vector_type(16))) _Float16 v16h;
typedef __attribute__((ext_vector_type(8)))  float    v8f;
typedef __attribute__((ext_vector_type(4)))  unsigned int v4u;
typedef __attribute__((ext_vector_type(8)))  unsigned int v8u;

#define N_BATCH 16
#define P_PTS   866
#define C_IN    1280
#define C_MID   512
#define HW      64
#define NOUT    10   // 2 + DEPTH
#define WROW    ((C_MID + 1) * NOUT)   // 5130 floats per (n,p) weight block

__device__ __forceinline__ float wave_sum(float v) {
#pragma unroll
  for (int o = 16; o > 0; o >>= 1) v += __shfl_xor(v, o, 32);
  return v;
}
__device__ __forceinline__ float wave_max(float v) {
#pragma unroll
  for (int o = 16; o > 0; o >>= 1) v = fmaxf(v, __shfl_xor(v, o, 32));
  return v;
}

// ---- prep: conv_w f32 -> f16 (same layout [c][ci]) ----
__global__ void k_prep_convw(const float* __restrict__ w, _Float16* __restrict__ o) {
  int i = blockIdx.x * 256 + threadIdx.x;
  if (i < C_MID * C_IN) o[i] = (_Float16)w[i];
}

// ---- prep: features (n,ci,x) f32 -> f16 transposed [n][x][ci] ----
__global__ void k_prep_feat(const float* __restrict__ f, _Float16* __restrict__ o) {
  int i = blockIdx.x * 256 + threadIdx.x;
  if (i < N_BATCH * HW * C_IN) {
    int ci = i % C_IN;
    int nx = i / C_IN;
    int x  = nx & (HW - 1);
    int n  = nx / HW;
    o[i] = (_Float16)f[((size_t)n * C_IN + ci) * HW + x];
  }
}

// ---- stage 1: f = silu(bn(conv_w @ features)), stored f16 as [n][x][c] ----
// 1 wave per (n, 16-channel tile). M=16 c, N=64 x (4 tiles), K=1280.
__global__ void __launch_bounds__(32)
k_conv(const _Float16* __restrict__ cwh, const _Float16* __restrict__ ft1,
       const float* __restrict__ g, const float* __restrict__ be,
       const float* __restrict__ mu, const float* __restrict__ var,
       _Float16* __restrict__ ft2) {
  int bid = blockIdx.x;
  int n = bid >> 5;            // 32 c-tiles
  int cbase = (bid & 31) * 16;
  int lane = threadIdx.x;
  int mo = lane & 15;          // A: M row ; B/C: N col
  int hi = lane >> 4;

  v8f acc[4] = {};
  const _Float16* fb = ft1 + (size_t)n * HW * C_IN;

  for (int kb = 0; kb < C_IN; kb += 32) {
    // A operand: conv_w tile, M=mo, halves i -> K = (i&7) + 16*(i>>3) + 8*hi
    v16h a;
    const _Float16* arow = cwh + (size_t)(cbase + mo) * C_IN + kb + hi * 8;
#pragma unroll
    for (int i = 0; i < 16; ++i) a[i] = arow[(i & 7) + ((i >> 3) << 4)];

    // Load all 4 B tiles first so the scheduler can stage s_wait_loadcnt.
    v16h bt[4];
#pragma unroll
    for (int t = 0; t < 4; ++t) {
      const _Float16* brow = fb + (size_t)(t * 16 + mo) * C_IN + kb + hi * 8;
#pragma unroll
      for (int j = 0; j < 8; ++j) { bt[t][2 * j] = brow[((2 * j) & 7) + ((j >> 2) << 4)]; bt[t][2 * j + 1] = brow[((2 * j) & 7) + ((j >> 2) << 4) + 1]; }
    }
#pragma unroll
    for (int t = 0; t < 4; ++t)
      acc[t] = __builtin_amdgcn_wmma_f32_16x16x32_f16(
          false, a, false, bt[t], (short)0, acc[t], false, false);
  }

  // BN affine + SiLU, store f16 [n][x][c]
#pragma unroll
  for (int j = 0; j < 8; ++j) {
    int c = cbase + j + 8 * hi;
    float sc = g[c] * rsqrtf(var[c] + 1e-3f);
    float sh = be[c] - mu[c] * sc;
#pragma unroll
    for (int t = 0; t < 4; ++t) {
      int x = t * 16 + mo;
      float v = acc[t][j] * sc + sh;
      v = v / (1.0f + __expf(-v));          // silu
      ft2[((size_t)n * HW + x) * C_MID + c] = (_Float16)v;
    }
  }
}

// ---- stage 2+3 fused: logits GEMM (WMMA) + softmaxes + decodes ----
// 1 wave per (n,p). M=16 (10 outputs padded), N=64 x (4 tiles), K=512.
// w_t (the HBM-critical 284MB stream) is DMA'd via the Tensor Data Mover:
// one tensor_load_to_lds per wave pulls the contiguous 20.5KB block into LDS,
// then the A operand is built from LDS with f32->f16 conversion per k-step.
__global__ void __launch_bounds__(32)
k_logits(const float* __restrict__ weights, const _Float16* __restrict__ ft2,
         float* __restrict__ out) {
  int bid = blockIdx.x;
  int n = bid / P_PTS;
  int p = bid - n * P_PTS;
  int lane = threadIdx.x;
  int mo = lane & 15;
  int hi = lane >> 4;

  const float* wb = weights + ((size_t)n * P_PTS + p) * WROW;
  const _Float16* fb = ft2 + (size_t)n * HW * C_MID;

  __shared__ float wRaw[WROW];            // 20,520 B : TDM destination
  __shared__ float lg[NOUT * HW];         //  2,560 B

  // ---- TDM: stage w_t block (contiguous 5130 f32) into LDS ----
  {
    unsigned lds_off = (unsigned)(unsigned long long)(const void*)&wRaw[0]; // LDS aperture truncates to offset
    unsigned long long ga = (unsigned long long)(const void*)wb;
    v4u g0;
    g0.x = 1u;                                            // count=1, user descriptor
    g0.y = lds_off;                                       // lds_addr
    g0.z = (unsigned)ga;                                  // global_addr[31:0]
    g0.w = (unsigned)((ga >> 32) & 0x1FFFFFFu) | (2u << 30); // global_addr[56:32] | type=2
    v8u g1;
    g1.s0 = 2u << 16;                                     // data_size = 4 bytes
    g1.s1 = ((unsigned)WROW & 0xFFFFu) << 16;             // tensor_dim0[15:0] @ bits63:48
    g1.s2 = ((unsigned)WROW >> 16) | (1u << 16);          // tensor_dim0 hi | tensor_dim1=1
    g1.s3 = ((unsigned)WROW & 0xFFFFu) << 16;             // tile_dim0 = 5130 @ bits127:112
    g1.s4 = 1u;                                           // tile_dim1 = 1 (tile_dim2 unused)
    g1.s5 = (unsigned)WROW;                               // tensor_dim0_stride[31:0]
    g1.s6 = 0u;
    g1.s7 = 0u;
    asm volatile("tensor_load_to_lds %0, %1"
                 :
                 : "s"(g0), "s"(g1)
                 : "memory");
    __builtin_amdgcn_s_wait_tensorcnt(0);                 // s_wait_tensorcnt 0x0
  }

  v8f acc[4] = {};
  v16h a = {};                       // rows 10..15 stay zero (padding lanes)
  for (int kb = 0; kb < C_MID; kb += 32) {
    // A operand from TDM-staged LDS: w_t[c][o] -> A[M=o][K=c], cvt f32->f16
    if (mo < NOUT) {
#pragma unroll
      for (int i = 0; i < 16; ++i) {
        int K = kb + (i & 7) + ((i >> 3) << 4) + hi * 8;
        a[i] = (_Float16)wRaw[K * NOUT + mo];
      }
    }
    // Load all 4 B tiles, prefetch next k-step, then 4 WMMAs.
    v16h bt[4];
#pragma unroll
    for (int t = 0; t < 4; ++t) {
      const _Float16* brow = fb + (size_t)(t * 16 + mo) * C_MID + kb + hi * 8;
#pragma unroll
      for (int j = 0; j < 8; ++j) { bt[t][2 * j] = brow[((2 * j) & 7) + ((j >> 2) << 4)]; bt[t][2 * j + 1] = brow[((2 * j) & 7) + ((j >> 2) << 4) + 1]; }
      if (kb + 32 < C_MID) __builtin_prefetch(brow + 32, 0, 0);  // global_prefetch_b8
    }
#pragma unroll
    for (int t = 0; t < 4; ++t)
      acc[t] = __builtin_amdgcn_wmma_f32_16x16x32_f16(
          false, a, false, bt[t], (short)0, acc[t], false, false);
  }

  // bias add + spill 10x64 logits to LDS (never touches HBM)
#pragma unroll
  for (int j = 0; j < 8; ++j) {
    int M = j + 8 * hi;
    if (M < NOUT) {
      float bias = wRaw[C_MID * NOUT + M];
#pragma unroll
      for (int t = 0; t < 4; ++t) lg[M * HW + t * 16 + mo] = acc[t][j] + bias;
    }
  }
  __syncthreads();

  // each lane handles x0=lane, x1=lane+32 ; coords scaled by 1/7 (linspace 0..1)
  int x0 = lane, x1 = lane + 32;
  const float inv7 = 1.0f / 7.0f;
  float fx0 = (x0 & 7) * inv7, fy0 = (x0 >> 3) * inv7;
  float fx1 = (x1 & 7) * inv7, fy1 = (x1 >> 3) * inv7;

  // channel 1: 2D softmax -> metric cx, cy
  float v10 = lg[HW + x0], v11 = lg[HW + x1];
  float m1 = wave_max(fmaxf(v10, v11));
  float e10 = __expf(v10 - m1), e11 = __expf(v11 - m1);
  float inv1 = 1.0f / wave_sum(e10 + e11);
  float cx = wave_sum(e10 * fx0 + e11 * fx1) * inv1;
  float cy = wave_sum(e10 * fy0 + e11 * fy1) * inv1;

  // channels 2..9: softmax over 512, decodes + uncertainty coupling
  float lm = -3.4e38f;
#pragma unroll
  for (int d = 0; d < 8; ++d)
    lm = fmaxf(lm, fmaxf(lg[(2 + d) * HW + x0], lg[(2 + d) * HW + x1]));
  float m2 = wave_max(lm);

  float se = 0.f, px = 0.f, py = 0.f, pz = 0.f, h0 = 0.f, h1 = 0.f;
#pragma unroll
  for (int d = 0; d < 8; ++d) {
    float ea = __expf(lg[(2 + d) * HW + x0] - m2);
    float eb = __expf(lg[(2 + d) * HW + x1] - m2);
    se += ea + eb;
    h0 += ea; h1 += eb;                 // heatmap2d partials (pre-norm)
    px += ea * fx0 + eb * fx1;
    py += ea * fy0 + eb * fy1;
    pz += (ea + eb) * (d * inv7);
  }
  float inv2 = 1.0f / wave_sum(se);
  float xd = wave_sum(px) * inv2;
  float yd = wave_sum(py) * inv2;
  float zd = wave_sum(pz) * inv2;
  float unc = wave_sum(lg[x0] * h0 + lg[x1] * h1) * inv2;
  float u = fmaxf(unc, 0.f) + log1pf(__expf(-fabsf(unc)));   // stable softplus

  if (lane == 0) {
    const size_t NP = (size_t)N_BATCH * P_PTS;
    size_t idx = (size_t)n * P_PTS + p;
    out[idx * 2 + 0] = xd;                       // coords2d
    out[idx * 2 + 1] = yd;
    float* o3 = out + NP * 2;                    // coords3d
    o3[idx * 3 + 0] = cx;
    o3[idx * 3 + 1] = cy;
    o3[idx * 3 + 2] = zd;
    out[NP * 5 + idx] = u;                       // uncertainties
  }
}

extern "C" void kernel_launch(void* const* d_in, const int* in_sizes, int n_in,
                              void* d_out, int out_size, void* d_ws, size_t ws_size,
                              hipStream_t stream) {
  const float* features = (const float*)d_in[0];
  const float* weights  = (const float*)d_in[1];
  const float* conv_w   = (const float*)d_in[2];
  const float* bn_gamma = (const float*)d_in[3];
  const float* bn_beta  = (const float*)d_in[4];
  const float* bn_mean  = (const float*)d_in[5];
  const float* bn_var   = (const float*)d_in[6];
  float* out = (float*)d_out;

  char* ws = (char*)d_ws;
  _Float16* ft1 = (_Float16*)(ws);                          // feat f16 [n][x][ci]  : 2,621,440 B
  _Float16* cwh = (_Float16*)(ws + 2621440);                // conv_w f16          : 1,310,720 B
  _Float16* ft2 = (_Float16*)(ws + 2621440 + 1310720);      // f f16 [n][x][c]     : 1,048,576 B

  int nfeat = N_BATCH * HW * C_IN;
  int ncw   = C_MID * C_IN;
  k_prep_feat<<<(nfeat + 255) / 256, 256, 0, stream>>>(features, ft1);
  k_prep_convw<<<(ncw + 255) / 256, 256, 0, stream>>>(conv_w, cwh);
  k_conv<<<N_BATCH * (C_MID / 16), 32, 0, stream>>>(cwh, ft1, bn_gamma, bn_beta,
                                                    bn_mean, bn_var, ft2);
  k_logits<<<N_BATCH * P_PTS, 32, 0, stream>>>(weights, ft2, out);
}
